// TwoSimplicialAttention_85667417686333
// MI455X (gfx1250) — matrix operand order repaired and checked by reading, not compiled, not run on hardware
//
#include <hip/hip_runtime.h>
#include <hip/hip_bf16.h>
#include <math.h>

typedef _Float16 h16;
typedef __attribute__((ext_vector_type(16))) _Float16 v16h;
typedef __attribute__((ext_vector_type(8)))  _Float16 v8h;
typedef __attribute__((ext_vector_type(8)))  float    v8f;
typedef int v4i __attribute__((vector_size(16)));   // matches builtin param type

#define B_    2
#define T_    192
#define C_    512
#define H_    8
#define DH_   64
#define PROJC (5*C_)   // 2560
// D^-0.5 * log2(e): softmax done in base-2 so exp2f lowers to bare v_exp_f32
#define QSCALE (0.125f * 1.4426950408889634f)

// ---------------------------------------------------------------------------
// gfx1250 async global->LDS path (ASYNCcnt-tracked), with safe fallback
// ---------------------------------------------------------------------------
#if defined(__has_builtin)
#  if __has_builtin(__builtin_amdgcn_global_load_async_to_lds_b128)
#    define HAVE_ASYNC_LDS 1
#  else
#    define HAVE_ASYNC_LDS 0
#  endif
#else
#  define HAVE_ASYNC_LDS 0
#endif

#if HAVE_ASYNC_LDS
typedef __attribute__((address_space(1))) v4i* as1_v4i_p;
typedef __attribute__((address_space(3))) v4i* as3_v4i_p;
#endif

static __device__ __forceinline__ void async_cp16(h16* ldst, const h16* gsrc){
#if HAVE_ASYNC_LDS
  __builtin_amdgcn_global_load_async_to_lds_b128(
      (as1_v4i_p)(gsrc), (as3_v4i_p)(ldst), 0, 0);
#else
  *(v8h*)ldst = *(const v8h*)gsrc;
#endif
}
static __device__ __forceinline__ void async_wait_all(){
#if HAVE_ASYNC_LDS
#  if defined(__has_builtin) && __has_builtin(__builtin_amdgcn_s_wait_asynccnt)
  __builtin_amdgcn_s_wait_asynccnt(0);
#  else
  asm volatile("s_wait_asynccnt 0x0" ::: "memory");
#  endif
#endif
}

// ---------------------------------------------------------------------------
// WMMA fragment helpers (contiguous K order in A and B, 16x16x32 f16)
// ---------------------------------------------------------------------------
static __device__ __forceinline__ v16h frag_cat(v8h a, v8h b){
  v16h r;
#pragma unroll
  for (int i=0;i<8;i++){ r[i]=a[i]; r[8+i]=b[i]; }
  return r;
}
// A-matrix fragment: lane group g=lane>>4; element e -> K=g*16+e (contiguous, matches B)
static __device__ __forceinline__ v16h ld_afrag(const h16* row, int g){
  v8h lo = *(const v8h*)(row + g*16);
  v8h hi = *(const v8h*)(row + g*16 + 8);
  return frag_cat(lo, hi);
}
// B-matrix fragment: lane holds fixed N, 16 contiguous K values starting at p
static __device__ __forceinline__ v16h ld_bfrag(const h16* p){
  v8h lo = *(const v8h*)(p);
  v8h hi = *(const v8h*)(p + 8);
  return frag_cat(lo, hi);
}
static __device__ __forceinline__ v8f wmma_f16(v16h a, v16h b, v8f c){
  return __builtin_amdgcn_wmma_f32_16x16x32_f16(false, a, false, b, (short)0, c,
                                                false, false);
}
static __device__ __forceinline__ v16h vmul(v16h a, v16h b){
  v16h r;
#pragma unroll
  for (int i=0;i<16;i++) r[i] = a[i]*b[i];
  return r;
}
static __device__ __forceinline__ v16h vscale(v16h a, h16 s){
  v16h r;
#pragma unroll
  for (int i=0;i<16;i++) r[i] = a[i]*s;
  return r;
}

// ---------------------------------------------------------------------------
// Generic 64x64-tile WMMA GEMM: C = A(f32, MxK) @ B(f32, KxN) + bias
// OUT_F16 ? write h16 to Ch : write float to Cf.   M,N %64==0, K%32==0.
// ---------------------------------------------------------------------------
template<bool OUT_F16>
__global__ __launch_bounds__(256)
void wmma_gemm_kernel(const float* __restrict__ A, const float* __restrict__ Bm,
                      const float* __restrict__ bias, float* __restrict__ Cf,
                      h16* __restrict__ Ch, int M, int N, int K)
{
  __shared__ h16 sA [64*32];   // [m][k] row-major
  __shared__ h16 sBt[64*32];   // [n][k] (transposed for B-fragment reads)
  const int tid  = threadIdx.x;
  const int wid  = tid>>5, lane = tid&31, g = lane>>4, ln = lane&15;
  const int m0 = blockIdx.y*64, n0 = blockIdx.x*64;
  const int it = wid>>1, jt = (wid&1)*2;   // wave owns 1 i-tile x 2 j-tiles
  v8f acc0 = {}; v8f acc1 = {};

  for (int k0=0; k0<K; k0+=32){
#pragma unroll
    for (int t=0;t<8;t++){
      int idx = tid + t*256;
      int i = idx>>5, k = idx&31;
      sA[idx] = (h16)A[(size_t)(m0+i)*K + (k0+k)];
    }
#pragma unroll
    for (int t=0;t<8;t++){
      int idx = tid + t*256;
      int k = idx>>6, n = idx&63;                 // coalesced over n
      sBt[n*32+k] = (h16)Bm[(size_t)(k0+k)*N + (n0+n)];
    }
    if (k0 + 32 < K){                             // gfx1250 global_prefetch_b8
      __builtin_prefetch(&A [(size_t)(m0+(tid>>3))*K + (k0+32)+(tid&7)*4], 0, 1);
      __builtin_prefetch(&Bm[(size_t)(k0+32+(tid>>6))*N + n0+(tid&63)],    0, 1);
    }
    __syncthreads();
    v16h a  = ld_afrag(sA + (it*16+ln)*32, g);
    v16h b0 = ld_bfrag(sBt + ( jt   *16+ln)*32 + g*16);
    v16h b1 = ld_bfrag(sBt + ((jt+1)*16+ln)*32 + g*16);
    acc0 = wmma_f16(a, b0, acc0);
    acc1 = wmma_f16(a, b1, acc1);
    __syncthreads();
  }
  const float bn0 = bias[n0 +  jt   *16 + ln];
  const float bn1 = bias[n0 + (jt+1)*16 + ln];
#pragma unroll
  for (int r=0;r<8;r++){
    int m  = m0 + it*16 + g*8 + r;                // C/D layout row
    int na = n0 +  jt   *16 + ln;
    int nb = n0 + (jt+1)*16 + ln;
    float va = acc0[r] + bn0, vb = acc1[r] + bn1;
    if constexpr (OUT_F16){
      Ch[(size_t)m*N + na] = (h16)va; Ch[(size_t)m*N + nb] = (h16)vb;
    } else {
      Cf[(size_t)m*N + na] = va;      Cf[(size_t)m*N + nb] = vb;
    }
  }
}

// ---------------------------------------------------------------------------
// Two-simplicial flash attention. One block = one (b,h) and one 64-query tile.
// proj layout per token (row of 2560 h16): [q | k1 | v1 | k2 | v2], each 512 =
// h*64+d. Online softmax (base-2) over the joint (j,kk) axis, streamed over kk.
// ---------------------------------------------------------------------------
__global__ __launch_bounds__(256)
void attn2s_kernel(const h16* __restrict__ proj, float* __restrict__ attout)
{
  extern __shared__ char smem[];
  h16*   sQ     = (h16*)smem;           // 64 x 64   (q * D^-0.5 * log2e)
  h16*   sK1    = sQ   + 64*64;         // 192 x 64
  h16*   sK2    = sK1  + 192*64;        // 192 x 64
  h16*   sV2    = sK2  + 192*64;        // 192 x 64
  h16*   sV1t   = sV2  + 192*64;        // 64 x 192  (v1 transposed: [d][j])
  h16*   sP     = sV1t + 64*192;        // 64 x 192  exp2(S - m)
  float* sS     = (float*)(sP + 64*192);// 64 x 192  raw logits for current kk
  float* sScale = sS + 64*192;          // 64        per-row rescale exp2(m_old-m_new)
  float* sL     = sScale + 64;          // 64        final softmax denominators

  const int tid = threadIdx.x;
  const int wid = tid>>5, lane = tid&31, g = lane>>4, ln = lane&15;
  const int b = blockIdx.z, h = blockIdx.y;
  const int i0 = blockIdx.x*64;

  // ---- stage q (scaled) and v1^T via VGPR path ----
#pragma unroll 1
  for (int t=0;t<16;t++){
    int idx = tid + t*256; int i = idx>>6, d = idx&63;
    float v = (float)proj[(size_t)(b*T_ + i0 + i)*PROJC + h*DH_ + d];
    sQ[idx] = (h16)(v * QSCALE);
  }
#pragma unroll 1
  for (int t=0;t<48;t++){
    int idx = tid + t*256; int j = idx>>6, d = idx&63;
    sV1t[d*192+j] = proj[(size_t)(b*T_ + j)*PROJC + h*DH_ + d + 2*C_];
  }
  // ---- stage k1/k2/v2 rows: 16B chunks, async to LDS when available ----
#pragma unroll 1
  for (int t=0;t<6;t++){
    int chunk = tid + t*256;            // 1536 chunks of 8 halves per matrix
    int j = chunk>>3, c8 = (chunk&7)*8; // row-major: lds half-offset = chunk*8
    size_t gbase = (size_t)(b*T_ + j)*PROJC + h*DH_ + c8;
    async_cp16(sK1 + chunk*8, proj + gbase + 1*C_);
    async_cp16(sK2 + chunk*8, proj + gbase + 3*C_);
    async_cp16(sV2 + chunk*8, proj + gbase + 4*C_);
  }
  async_wait_all();
  __syncthreads();

  const int it  = wid & 3;        // i-tile owned for S-compute and PV-acc
  const int jt0 = (wid>>2)*6;     // j-tiles 0-5 or 6-11 for S-compute
  const int dt0 = (wid>>2)*2;     // d-tiles owned for PV-acc

  const h16* qrow = sQ + (it*16 + ln)*DH_;
  const v16h aq0 = ld_afrag(qrow,      g);   // q fragment, K = d 0..31
  const v16h aq1 = ld_afrag(qrow + 32, g);   // q fragment, K = d 32..63
  v8f acc0 = {}; v8f acc1 = {};
  float m_i = -INFINITY, l_i = 0.f;
  const int srow = tid >> 2, spart = tid & 3;  // softmax: 4 lanes per row

  for (int kk=0; kk<T_; kk++){
    // A = q_i * k2_kk (elementwise in A-fragment layout; same K map per lane)
    const h16* k2row = sK2 + kk*DH_;
    v16h a0 = vmul(aq0, ld_afrag(k2row,      g));
    v16h a1 = vmul(aq1, ld_afrag(k2row + 32, g));
    // S[i,j] = (q_i*k2_kk) . k1_j   -> 6 j-tiles per wave
#pragma unroll
    for (int t=0;t<6;t++){
      int jt = jt0 + t;
      const h16* kb = sK1 + (jt*16 + ln)*DH_ + g*16;
      v8f s = {};
      s = wmma_f16(a0, ld_bfrag(kb),      s);
      s = wmma_f16(a1, ld_bfrag(kb + 32), s);
      float* so = sS + (it*16 + g*8)*192 + jt*16 + ln;
#pragma unroll
      for (int r=0;r<8;r++) so[r*192] = s[r];
    }
    __syncthreads();
    // ---- online softmax, base-2 (4 consecutive lanes per row -> shfl reduce)
    {
      const float* sr = sS + srow*192 + spart*48;
      float rm = sr[0];
#pragma unroll 1
      for (int j=1;j<48;j++) rm = fmaxf(rm, sr[j]);
      rm = fmaxf(rm, __shfl_xor(rm, 1, 32));
      rm = fmaxf(rm, __shfl_xor(rm, 2, 32));
      float mnew = fmaxf(m_i, rm);
      float sc   = exp2f(m_i - mnew);           // bare v_exp_f32
      h16* pr = sP + srow*192 + spart*48;
      float rs = 0.f;
#pragma unroll 1
      for (int j=0;j<48;j++){
        float e = exp2f(sr[j] - mnew);
        rs += e;
        pr[j] = (h16)e;
      }
      rs += __shfl_xor(rs, 1, 32);
      rs += __shfl_xor(rs, 2, 32);
      l_i = l_i*sc + rs;
      m_i = mnew;
      if (spart==0) sScale[srow] = sc;
    }
    __syncthreads();
    // ---- rescale accumulators by exp2(m_old - m_new) per row ----
    float scr[8];
#pragma unroll
    for (int r=0;r<8;r++) scr[r] = sScale[it*16 + g*8 + r];
#pragma unroll
    for (int r=0;r<8;r++){ acc0[r]*=scr[r]; acc1[r]*=scr[r]; }
    // ---- acc += P @ (v1 * v2[kk,d]); v2 factor is per-lane (fixed d per lane)
    h16 v2a = sV2[kk*DH_ +  dt0   *16 + ln];
    h16 v2b = sV2[kk*DH_ + (dt0+1)*16 + ln];
#pragma unroll
    for (int c=0;c<6;c++){
      const h16* prow = sP + (it*16 + ln)*192 + c*32;
      v16h ap  = ld_afrag(prow, g);
      v16h bv0 = vscale(ld_bfrag(sV1t + ( dt0   *16+ln)*192 + c*32 + g*16), v2a);
      v16h bv1 = vscale(ld_bfrag(sV1t + ((dt0+1)*16+ln)*192 + c*32 + g*16), v2b);
      acc0 = wmma_f16(ap, bv0, acc0);
      acc1 = wmma_f16(ap, bv1, acc1);
    }
    // no trailing barrier needed: next S-write (sS) is disjoint from PV reads
    // (sP/sScale), and next softmax writes happen only after the next barrier.
  }

  if (spart==0) sL[srow] = l_i;
  __syncthreads();
  float inv[8];
#pragma unroll
  for (int r=0;r<8;r++) inv[r] = 1.f / sL[it*16 + g*8 + r];
#pragma unroll
  for (int r=0;r<8;r++){
    int i = i0 + it*16 + g*8 + r;
    size_t base = (size_t)(b*T_ + i)*C_ + h*DH_;   // heads merged: (B,T,H*D)
    attout[base +  dt0   *16 + ln] = acc0[r]*inv[r];
    attout[base + (dt0+1)*16 + ln] = acc1[r]*inv[r];
  }
}

// ---------------------------------------------------------------------------
extern "C" void kernel_launch(void* const* d_in, const int* in_sizes, int n_in,
                              void* d_out, int out_size, void* d_ws, size_t ws_size,
                              hipStream_t stream)
{
  (void)in_sizes; (void)n_in; (void)out_size; (void)ws_size;
  const float* x     = (const float*)d_in[0];   // (B,T,C)
  const float* W_in  = (const float*)d_in[1];   // (C, 5*H*D)
  const float* b_in  = (const float*)d_in[2];   // (5*H*D)
  const float* W_out = (const float*)d_in[3];   // (H*D, C)
  const float* b_out = (const float*)d_in[4];   // (C)

  h16*   proj = (h16*)d_ws;                                      // 384x2560 f16
  float* att  = (float*)((char*)d_ws + (size_t)(B_*T_)*PROJC*2); // 384x512 f32
  float* y    = (float*)d_out;

  dim3 blk(256);
  // 1) proj = x @ W_in + b_in  (f16 out)
  wmma_gemm_kernel<true><<<dim3(PROJC/64, (B_*T_)/64), blk, 0, stream>>>(
      x, W_in, b_in, nullptr, proj, B_*T_, PROJC, C_);

  // 2) two-simplicial flash attention -> att (f32, heads merged)
  size_t smem = (size_t)(64*64 + 3*192*64 + 64*192 + 64*192)*sizeof(h16)
              + (size_t)(64*192 + 128)*sizeof(float);
  attn2s_kernel<<<dim3(T_/64, H_, B_), blk, smem, stream>>>(proj, att);

  // 3) y = att @ W_out + b_out (f32 out)
  wmma_gemm_kernel<false><<<dim3(C_/64, (B_*T_)/64), blk, 0, stream>>>(
      att, W_out, b_out, y, nullptr, B_*T_, C_, C_);
}
